// DeformableAlignment_39754217291858
// MI455X (gfx1250) — matrix operand order repaired and checked by reading, not compiled, not run on hardware
//
#include <hip/hip_runtime.h>

#define BATCH 4
#define CIN   128
#define COUT  128
#define H     128
#define W     128
#define HW    (H * W)
#define CCOND 261
#define LDSB_STRIDE 40   // bf16 elements per LDS B row (16B-aligned fragment reads)

typedef __bf16 bf16_t;
typedef __attribute__((ext_vector_type(16))) __bf16 v16bf;
typedef __attribute__((ext_vector_type(8)))  __bf16 v8bf;
typedef __attribute__((ext_vector_type(8)))  float  v8f;
typedef int v4i __attribute__((vector_size(16)));
typedef __attribute__((address_space(1))) v4i gv4i;   // global int4
typedef __attribute__((address_space(3))) v4i lv4i;   // LDS int4

#if defined(__has_builtin)
#if __has_builtin(__builtin_amdgcn_global_load_async_to_lds_b128) && \
    __has_builtin(__builtin_amdgcn_s_wait_asynccnt)
#define HAS_ASYNC_LDS 1
#endif
#endif

#ifdef HAS_ASYNC_LDS
#define ASYNC_COPY_B128(g, l)                                                  \
  __builtin_amdgcn_global_load_async_to_lds_b128((gv4i*)(g), (lv4i*)(l), 0, 0)
#define WAIT_ASYNC(n) __builtin_amdgcn_s_wait_asynccnt(n)
#else
#define ASYNC_COPY_B128(g, l) (*(v8bf*)(l) = *(const v8bf*)(g))
#define WAIT_ASYNC(n) ((void)0)
#endif

static __device__ __forceinline__ unsigned short f2bfu(float f) {
  union { float f; unsigned u; } v; v.f = f;
  return (unsigned short)((v.u + 0x7FFFu + ((v.u >> 16) & 1u)) >> 16);  // RNE
}

// ---------------------------------------------------------------------------
// Weight prep: fp32 [Cout][K] (K = Cin*9, row-major from OIHW) -> bf16
// padded [CoutPad][Kpad], zero-filled pad (so padded WMMA tiles are exact).
// ---------------------------------------------------------------------------
__global__ void wprep_kernel(const float* __restrict__ w, bf16_t* __restrict__ out,
                             int Cout, int K, int CoutPad, int Kpad) {
  int id = blockIdx.x * 256 + threadIdx.x;
  int total = CoutPad * Kpad;
  if (id >= total) return;
  int co = id / Kpad, k = id - co * Kpad;
  float v = (co < Cout && k < K) ? w[(size_t)co * K + k] : 0.0f;
  unsigned short h = f2bfu(v);
  __builtin_memcpy(&out[id], &h, 2);
}

// ---------------------------------------------------------------------------
// Issue async copy of one wave's 16x32 bf16 A-tile (1KB) into its private
// LDS slot: 2 x B128 issues; lane l covers row=(l>>2)+8j, 16B segment l&3.
// Producer wave == consumer wave, so only s_wait_asynccnt is needed.
// ---------------------------------------------------------------------------
static __device__ __forceinline__ void issue_a_tile(const bf16_t* __restrict__ wbf,
                                                    bf16_t* lds_slot, int row0,
                                                    int Kpad, int kk, int lane) {
  int row = lane >> 2, seg = lane & 3;
  #pragma unroll
  for (int j = 0; j < 2; ++j) {
    int r = row + j * 8;
    const bf16_t* g = wbf + (size_t)(row0 + r) * Kpad + kk + seg * 8;
    bf16_t* l = lds_slot + r * 32 + seg * 8;
    ASYNC_COPY_B128(g, l);
  }
}

// ---------------------------------------------------------------------------
// 3x3 SAME conv as implicit GEMM, V_WMMA_F32_16X16X32_BF16.
// Block = 256 thr (8 waves). Block tile: 32 pixels x up to 128 couts.
// Each wave: one 16-cout row, two 16-pixel B tiles -> 2 WMMA per K-chunk
// reusing one A fragment, which is double-buffered via async-to-LDS.
// ---------------------------------------------------------------------------
__global__ __launch_bounds__(256)
void conv3x3_wmma_kernel(const float* __restrict__ in, float* __restrict__ out,
                         const bf16_t* __restrict__ wbf, const float* __restrict__ bias,
                         int Cin, int Kpad, int Cout, int nCoutTiles, int act) {
  __shared__ __align__(16) bf16_t lds_b[2][16 * LDSB_STRIDE];
  __shared__ __align__(16) bf16_t lds_a[2 * 8 * 512];   // [buf][wave] 16x32 tiles

  const int tid  = threadIdx.x;
  const int lane = tid & 31;
  const int wave = tid >> 5;
  const int half = lane >> 4;
  const int l16  = lane & 15;

  const int tile = blockIdx.x;                    // 4*128*4 blocks
  const int xt = (tile & 3) << 5;                 // 32-pixel x segment
  const int y  = (tile >> 2) & (H - 1);
  const int b  = tile >> 9;

  const int K     = Cin * 9;
  const int cout0 = wave * 16;
  const bool act_wave = (wave < nCoutTiles);

  v8f acc0, acc1;
  #pragma unroll
  for (int i = 0; i < 8; ++i) {
    int co = cout0 + half * 8 + i;
    float bv = (act_wave && co < Cout) ? bias[co] : 0.0f;
    acc0[i] = bv; acc1[i] = bv;
  }

  const int px32 = tid & 31;                      // build: pixel in segment
  const int kb   = (tid >> 5) * 4;                // build: 4 consecutive k
  const int bt   = px32 >> 4;                     // B tile 0/1
  const int p16  = px32 & 15;

  int buf = 0;
  if (act_wave) issue_a_tile(wbf, &lds_a[wave * 512], cout0, Kpad, 0, lane);

  const int nChunks = Kpad / 32;
  for (int ci = 0; ci < nChunks; ++ci) {
    const int kk = ci * 32;
    if (act_wave && ci + 1 < nChunks)
      issue_a_tile(wbf, &lds_a[(buf ^ 1) * 8 * 512 + wave * 512],
                   cout0, Kpad, kk + 32, lane);

    __syncthreads();
    // Build im2col B tiles [2][16 px][32 k] (bf16), 4 k per thread, packed.
    unsigned short h[4];
    #pragma unroll
    for (int j = 0; j < 4; ++j) {
      int kg = kk + kb + j;
      float v = 0.0f;
      if (kg < K) {
        int cin = kg / 9;
        int tap = kg - cin * 9;
        int ky  = tap / 3;
        int yy  = y - 1 + ky;
        int xx  = xt + px32 - 1 + (tap - ky * 3);
        if ((unsigned)yy < (unsigned)H && (unsigned)xx < (unsigned)W)
          v = in[(((size_t)b * Cin + cin) * H + yy) * W + xx];
      }
      h[j] = f2bfu(v);
    }
    uint2 pk;
    pk.x = (unsigned)h[0] | ((unsigned)h[1] << 16);
    pk.y = (unsigned)h[2] | ((unsigned)h[3] << 16);
    *(uint2*)&lds_b[bt][p16 * LDSB_STRIDE + kb] = pk;
    __syncthreads();

    if (act_wave) {
      if (ci + 1 < nChunks) WAIT_ASYNC(2); else WAIT_ASYNC(0);
      // A fragment from this wave's LDS slot (ISA 16-bit A 16x32 layout)
      const char* abase = (const char*)&lds_a[buf * 8 * 512 + wave * 512];
      const v8bf* ap = (const v8bf*)(abase + l16 * 64 + half * 16);
      v8bf a_lo = ap[0], a_hi = ap[2];
      v16bf a;
      #pragma unroll
      for (int i = 0; i < 8; ++i) { a[i] = a_lo[i]; a[8 + i] = a_hi[i]; }
      // B fragments: N = l16, K = half*8 + e (e<8), 16 + half*8 + (e-8) (e>=8)
      const v8bf* br0 = (const v8bf*)(&lds_b[0][l16 * LDSB_STRIDE + half * 8]);
      const v8bf* br1 = (const v8bf*)(&lds_b[1][l16 * LDSB_STRIDE + half * 8]);
      v8bf b0l = br0[0], b0h = br0[2], b1l = br1[0], b1h = br1[2];
      v16bf bm0, bm1;
      #pragma unroll
      for (int i = 0; i < 8; ++i) {
        bm0[i] = b0l[i]; bm0[8 + i] = b0h[i];
        bm1[i] = b1l[i]; bm1[8 + i] = b1h[i];
      }
      acc0 = __builtin_amdgcn_wmma_f32_16x16x32_bf16(false, a, false, bm0,
                                                     (short)0, acc0, false, false);
      acc1 = __builtin_amdgcn_wmma_f32_16x16x32_bf16(false, a, false, bm1,
                                                     (short)0, acc1, false, false);
    }
    buf ^= 1;
  }

  if (act_wave) {
    #pragma unroll
    for (int i = 0; i < 8; ++i) {
      int co = cout0 + half * 8 + i;               // D: M = i + 8*half, N = l16
      if (co < Cout) {
        size_t rowp = (((size_t)b * Cout + co) * H + y) * W + xt + l16;
        float f0 = acc0[i], f1 = acc1[i];
        if (act) {
          f0 = (f0 >= 0.0f) ? f0 : 0.1f * f0;
          f1 = (f1 >= 0.0f) ? f1 : 0.1f * f1;
        }
        out[rowp]      = f0;
        out[rowp + 16] = f1;
      }
    }
  }
}

// ---------------------------------------------------------------------------
// head[B,27,H,W] + flow[B,2,H,W] -> offmask[B,27,H,W]
// ---------------------------------------------------------------------------
__global__ void offmask_kernel(const float* __restrict__ head,
                               const float* __restrict__ flow,
                               float* __restrict__ om) {
  int p = blockIdx.x * 256 + threadIdx.x;
  if (p >= BATCH * HW) return;
  int b = p / HW, r = p - b * HW;
  float fx = flow[((size_t)b * 2 + 0) * HW + r];
  float fy = flow[((size_t)b * 2 + 1) * HW + r];
  #pragma unroll
  for (int c = 0; c < 18; ++c) {
    float h = head[((size_t)b * 27 + c) * HW + r];
    om[((size_t)b * 27 + c) * HW + r] = 3.0f * tanhf(h) + ((c & 1) ? fx : fy);
  }
  #pragma unroll
  for (int c = 18; c < 27; ++c) {
    float h = head[((size_t)b * 27 + c) * HW + r];
    om[((size_t)b * 27 + c) * HW + r] = 1.0f / (1.0f + __expf(-h));
  }
}

// ---------------------------------------------------------------------------
// Modulated deformable conv as implicit GEMM (same skeleton as the convs).
// Stage 1: per (tap, pixel) record floor coords + 4 corner weights (x mask,
// x validity) in LDS. Stage 2: B-tile built by 4-corner fp32 gather/blend.
// ---------------------------------------------------------------------------
__global__ __launch_bounds__(256)
void deform_wmma_kernel(const float* __restrict__ x, const float* __restrict__ om,
                        const bf16_t* __restrict__ wbf, const float* __restrict__ bias,
                        float* __restrict__ out) {
  __shared__ __align__(16) bf16_t lds_b[2][16 * LDSB_STRIDE];
  __shared__ __align__(16) bf16_t lds_a[2 * 8 * 512];
  __shared__ int   s_y0[288];
  __shared__ int   s_x0[288];
  __shared__ __align__(16) float s_w[288 * 4];

  const int tid  = threadIdx.x;
  const int lane = tid & 31;
  const int wave = tid >> 5;
  const int half = lane >> 4;
  const int l16  = lane & 15;

  const int tile = blockIdx.x;
  const int xt = (tile & 3) << 5;
  const int y  = (tile >> 2) & (H - 1);
  const int b  = tile >> 9;

  // Stage 1: 9 taps x 32 pixels = 288 sampling records
  for (int e = tid; e < 288; e += 256) {
    int tap = e >> 5;
    int p   = e & 31;
    int xx  = xt + p;
    size_t base = (size_t)b * 27 * HW + (size_t)y * W + xx;
    float oy = om[base + (size_t)(2 * tap) * HW];
    float ox = om[base + (size_t)(2 * tap + 1) * HW];
    float m  = om[base + (size_t)(18 + tap) * HW];
    float py = (float)(y - 1 + tap / 3) + oy;
    float px = (float)(xx - 1 + (tap % 3)) + ox;
    float fy0 = floorf(py), fx0 = floorf(px);
    int y0 = (int)fy0, x0 = (int)fx0;
    float fy = py - fy0, fx = px - fx0;
    float vy0 = (y0 >= 0     && y0 <= H - 1) ? 1.0f : 0.0f;
    float vy1 = (y0 + 1 >= 0 && y0 + 1 <= H - 1) ? 1.0f : 0.0f;
    float vx0 = (x0 >= 0     && x0 <= W - 1) ? 1.0f : 0.0f;
    float vx1 = (x0 + 1 >= 0 && x0 + 1 <= W - 1) ? 1.0f : 0.0f;
    s_y0[e] = y0; s_x0[e] = x0;
    s_w[e * 4 + 0] = m * (1.0f - fy) * (1.0f - fx) * vy0 * vx0;
    s_w[e * 4 + 1] = m * (1.0f - fy) * fx          * vy0 * vx1;
    s_w[e * 4 + 2] = m * fy * (1.0f - fx)          * vy1 * vx0;
    s_w[e * 4 + 3] = m * fy * fx                   * vy1 * vx1;
  }

  const int cout0 = wave * 16;
  v8f acc0, acc1;
  #pragma unroll
  for (int i = 0; i < 8; ++i) {
    float bv = bias[cout0 + half * 8 + i];
    acc0[i] = bv; acc1[i] = bv;
  }

  const int px32 = tid & 31;
  const int kb   = (tid >> 5) * 4;
  const int bt   = px32 >> 4;
  const int p16  = px32 & 15;

  int buf = 0;
  issue_a_tile(wbf, &lds_a[wave * 512], cout0, CIN * 9, 0, lane);

  const int nChunks = (CIN * 9) / 32;             // 36
  for (int ci = 0; ci < nChunks; ++ci) {
    const int kk = ci * 32;
    if (ci + 1 < nChunks)
      issue_a_tile(wbf, &lds_a[(buf ^ 1) * 8 * 512 + wave * 512],
                   cout0, CIN * 9, kk + 32, lane);

    __syncthreads();                               // covers stage-1 on 1st iter
    unsigned short h[4];
    #pragma unroll
    for (int j = 0; j < 4; ++j) {
      int kg = kk + kb + j;
      int cin = kg / 9;
      int tap = kg - cin * 9;
      int sidx = tap * 32 + px32;
      int y0 = s_y0[sidx], x0 = s_x0[sidx];
      int yi0 = min(max(y0, 0), H - 1), yi1 = min(max(y0 + 1, 0), H - 1);
      int xi0 = min(max(x0, 0), W - 1), xi1 = min(max(x0 + 1, 0), W - 1);
      float4 wv = *(const float4*)&s_w[sidx * 4];
      const float* img = x + ((size_t)b * CIN + cin) * HW;
      float v = wv.x * img[yi0 * W + xi0] + wv.y * img[yi0 * W + xi1]
              + wv.z * img[yi1 * W + xi0] + wv.w * img[yi1 * W + xi1];
      h[j] = f2bfu(v);
    }
    uint2 pk;
    pk.x = (unsigned)h[0] | ((unsigned)h[1] << 16);
    pk.y = (unsigned)h[2] | ((unsigned)h[3] << 16);
    *(uint2*)&lds_b[bt][p16 * LDSB_STRIDE + kb] = pk;
    __syncthreads();

    if (ci + 1 < nChunks) WAIT_ASYNC(2); else WAIT_ASYNC(0);
    const char* abase = (const char*)&lds_a[buf * 8 * 512 + wave * 512];
    const v8bf* ap = (const v8bf*)(abase + l16 * 64 + half * 16);
    v8bf a_lo = ap[0], a_hi = ap[2];
    v16bf a;
    #pragma unroll
    for (int i = 0; i < 8; ++i) { a[i] = a_lo[i]; a[8 + i] = a_hi[i]; }
    const v8bf* br0 = (const v8bf*)(&lds_b[0][l16 * LDSB_STRIDE + half * 8]);
    const v8bf* br1 = (const v8bf*)(&lds_b[1][l16 * LDSB_STRIDE + half * 8]);
    v8bf b0l = br0[0], b0h = br0[2], b1l = br1[0], b1h = br1[2];
    v16bf bm0, bm1;
    #pragma unroll
    for (int i = 0; i < 8; ++i) {
      bm0[i] = b0l[i]; bm0[8 + i] = b0h[i];
      bm1[i] = b1l[i]; bm1[8 + i] = b1h[i];
    }
    acc0 = __builtin_amdgcn_wmma_f32_16x16x32_bf16(false, a, false, bm0,
                                                   (short)0, acc0, false, false);
    acc1 = __builtin_amdgcn_wmma_f32_16x16x32_bf16(false, a, false, bm1,
                                                   (short)0, acc1, false, false);
    buf ^= 1;
  }

  #pragma unroll
  for (int i = 0; i < 8; ++i) {
    int co = cout0 + half * 8 + i;
    size_t rowp = (((size_t)b * COUT + co) * H + y) * W + xt + l16;
    out[rowp]      = acc0[i];
    out[rowp + 16] = acc1[i];
  }
}

// ---------------------------------------------------------------------------
extern "C" void kernel_launch(void* const* d_in, const int* in_sizes, int n_in,
                              void* d_out, int out_size, void* d_ws, size_t ws_size,
                              hipStream_t stream) {
  (void)in_sizes; (void)n_in; (void)out_size; (void)ws_size;
  const float* x    = (const float*)d_in[0];
  const float* cond = (const float*)d_in[1];
  const float* flow = (const float*)d_in[2];
  const float* w1 = (const float*)d_in[3];  const float* b1 = (const float*)d_in[4];
  const float* w2 = (const float*)d_in[5];  const float* b2 = (const float*)d_in[6];
  const float* w3 = (const float*)d_in[7];  const float* b3 = (const float*)d_in[8];
  const float* w4 = (const float*)d_in[9];  const float* b4 = (const float*)d_in[10];
  const float* wd = (const float*)d_in[11]; const float* bd = (const float*)d_in[12];

  char* p = (char*)d_ws;
  auto carve = [&](size_t bytes) -> char* {
    char* r = p; p += (bytes + 255) & ~(size_t)255; return r;
  };
  float*  bufA = (float*)carve((size_t)BATCH * COUT * HW * 4);  // f1 / f3
  float*  bufB = (float*)carve((size_t)BATCH * COUT * HW * 4);  // f2
  float*  head = (float*)carve((size_t)BATCH * 27 * HW * 4);
  float*  om   = (float*)carve((size_t)BATCH * 27 * HW * 4);
  bf16_t* wb1  = (bf16_t*)carve((size_t)128 * 2368 * 2);        // K=2349 -> pad 2368
  bf16_t* wb2  = (bf16_t*)carve((size_t)128 * 1152 * 2);
  bf16_t* wb3  = (bf16_t*)carve((size_t)128 * 1152 * 2);
  bf16_t* wb4  = (bf16_t*)carve((size_t)32  * 1152 * 2);        // 27 -> pad 32 couts
  bf16_t* wbD  = (bf16_t*)carve((size_t)128 * 1152 * 2);

  auto nb = [](int n) { return (n + 255) / 256; };
  wprep_kernel<<<nb(128 * 2368), 256, 0, stream>>>(w1, wb1, 128, 2349, 128, 2368);
  wprep_kernel<<<nb(128 * 1152), 256, 0, stream>>>(w2, wb2, 128, 1152, 128, 1152);
  wprep_kernel<<<nb(128 * 1152), 256, 0, stream>>>(w3, wb3, 128, 1152, 128, 1152);
  wprep_kernel<<<nb(32  * 1152), 256, 0, stream>>>(w4, wb4, 27, 1152, 32, 1152);
  wprep_kernel<<<nb(128 * 1152), 256, 0, stream>>>(wd, wbD, 128, 1152, 128, 1152);

  dim3 grid(BATCH * H * (W / 32));  // 2048 workgroups, 32-pixel segments
  conv3x3_wmma_kernel<<<grid, 256, 0, stream>>>(cond, bufA, wb1, b1, CCOND, 2368, 128, 8, 1);
  conv3x3_wmma_kernel<<<grid, 256, 0, stream>>>(bufA, bufB, wb2, b2, 128, 1152, 128, 8, 1);
  conv3x3_wmma_kernel<<<grid, 256, 0, stream>>>(bufB, bufA, wb3, b3, 128, 1152, 128, 8, 1);
  conv3x3_wmma_kernel<<<grid, 256, 0, stream>>>(bufA, head, wb4, b4, 128, 1152, 27, 2, 0);
  offmask_kernel<<<nb(BATCH * HW), 256, 0, stream>>>(head, flow, om);
  deform_wmma_kernel<<<grid, 256, 0, stream>>>(x, om, wbD, bd, (float*)d_out);
}
